// QTIPLinearTCQ_6030134083832
// MI455X (gfx1250) — matrix operand order repaired and checked by reading, not compiled, not run on hardware
//
#include <hip/hip_runtime.h>
#include <hip/hip_bf16.h>

// QTIP trellis-quantized linear:  y[8192,4096] = x[8192,4096] @ W[4096,4096]^T
// Stage 1: decode trellis -> W bf16 hi/lo planes (m-major) in ws (64MB, L2-resident).
// Stage 2: split x -> bf16 hi/lo planes in ws (134MB) -- one-shot, memory bound.
// Stage 3: pure-WMMA GEMM, 3-term bf16 split (~f32 accuracy):
//          y = xhi*Whi + xlo*Whi + xhi*Wlo   (v_wmma_f32_16x16x32_bf16)
//          wave tile 64n x 32m -> 24 WMMA : 24 B128 loads per k-chunk.
// Fallback: if ws too small for x planes, GEMM converts x in-register.

#define MDIM 4096      // output features (rows of W)
#define KDIM 4096      // reduction dim
#define NROWS 8192     // BATCH*SEQ
#define NT_TILES 65536 // (M/16)*(K/16)

typedef __attribute__((ext_vector_type(16))) __bf16 v16bf;
typedef __attribute__((ext_vector_type(8)))  __bf16 v8bf;
typedef __attribute__((ext_vector_type(8)))  float  v8f;
typedef __attribute__((ext_vector_type(8)))  unsigned short v8us;

union Vb { v16bf v; unsigned short u[16]; v8bf h[2]; };
union Hs { v8us v; unsigned short u[8]; };

__device__ __forceinline__ unsigned short bf16_rne(float f) {
  unsigned u = __float_as_uint(f);
  unsigned r = u + 0x7FFFu + ((u >> 16) & 1u);   // round-to-nearest-even
  return (unsigned short)(r >> 16);
}

// ---------------------------------------------------------------------------
// Stage 1: decode. One thread per (tile, t-step). 65536*128 = 8.4M threads.
// code(t) = 9-bit MSB-first field at ring bit offset 4t+7 (mod 512).
// Store W m-major (W[m][k]) so GEMM B fragments are contiguous per lane.
// ---------------------------------------------------------------------------
__global__ __launch_bounds__(256) void qtip_decode(const int* __restrict__ trellis,
                                                   const float* __restrict__ tlut,
                                                   unsigned short* __restrict__ Whi,
                                                   unsigned short* __restrict__ Wlo)
{
  unsigned gid = blockIdx.x * 256u + threadIdx.x;
  if (gid >= (unsigned)NT_TILES * 128u) return;
  unsigned nt = gid >> 7;
  unsigned t  = gid & 127u;

  const int* tw = trellis + (size_t)nt * 32;     // 32 words, 16 payload bits each
  unsigned p  = 4u * t + 7u;                     // start bit of 9-bit code
  unsigned b0 = p & 511u;                        // ring wrap (tail-biting)
  unsigned w0 = b0 >> 4;
  unsigned o  = b0 & 15u;
  unsigned wa = (unsigned)tw[w0] & 0xFFFFu;
  unsigned wb = (unsigned)tw[(w0 + 1u) & 31u] & 0xFFFFu;
  unsigned v  = (wa << 16) | wb;
  unsigned code = (v >> (23u - o)) & 0x1FFu;     // 9 bits, MSB-first

  float f0 = tlut[code * 2u];
  float f1 = tlut[code * 2u + 1u];

  unsigned fl = 2u * t;                          // flat position in 16x16 tile
  unsigned r  = fl >> 4, c = fl & 15u;
  unsigned m  = (nt >> 8) * 16u + r;             // tile_row = nt / (K/16)
  unsigned k  = (nt & 255u) * 16u + c;           // tile_col = nt % (K/16)

  unsigned short h0 = bf16_rne(f0);
  unsigned short l0 = bf16_rne(f0 - __uint_as_float((unsigned)h0 << 16));
  unsigned short h1 = bf16_rne(f1);
  unsigned short l1 = bf16_rne(f1 - __uint_as_float((unsigned)h1 << 16));

  size_t b_0 = (size_t)m * KDIM + k;             // W[m][k] (m-major)
  size_t b_1 = (size_t)m * KDIM + k + 1u;
  Whi[b_0] = h0; Whi[b_1] = h1;
  Wlo[b_0] = l0; Wlo[b_1] = l1;
}

// ---------------------------------------------------------------------------
// Stage 2: one-shot x split. 8 elements/thread, vectorized 16B stores.
// ---------------------------------------------------------------------------
__global__ __launch_bounds__(256) void x_split(const float* __restrict__ x,
                                               unsigned short* __restrict__ Xhi,
                                               unsigned short* __restrict__ Xlo)
{
  size_t base = ((size_t)blockIdx.x * 256u + threadIdx.x) * 8u;
  if (base >= (size_t)NROWS * KDIM) return;
  const float4* xp = (const float4*)(x + base);
  float4 a = xp[0], b = xp[1];
  float f[8] = {a.x, a.y, a.z, a.w, b.x, b.y, b.z, b.w};
  Hs hi, lo;
#pragma unroll
  for (int j = 0; j < 8; ++j) {
    unsigned short h = bf16_rne(f[j]);
    hi.u[j] = h;
    lo.u[j] = bf16_rne(f[j] - __uint_as_float((unsigned)h << 16));
  }
  *(v8us*)(Xhi + base) = hi.v;
  *(v8us*)(Xlo + base) = lo.v;
}

// ---------------------------------------------------------------------------
// Stage 3 (fast path): pure-WMMA GEMM from pre-split bf16 planes.
// Block = 256 thr (8 waves): 2 n-waves x 4 m-waves -> 128n x 128m block tile.
// Wave tile = 64n x 32m (4x2 D tiles): per k-chunk of 32,
//   24x global_load_b128 and 24x v_wmma_f32_16x16x32_bf16 (ratio 1.0),
//   B fragments reused 4x, term-major order => 8 independent WMMAs between
//   accumulator reuses.
// ---------------------------------------------------------------------------
__global__ __launch_bounds__(256) void qtip_gemm_split(const unsigned short* __restrict__ Xhi,
                                                       const unsigned short* __restrict__ Xlo,
                                                       const unsigned short* __restrict__ WhiU,
                                                       const unsigned short* __restrict__ WloU,
                                                       float* __restrict__ y)
{
  const int lane  = threadIdx.x & 31;
  const int wave  = threadIdx.x >> 5;
  const int wn    = wave & 1;                  // 2 waves along n
  const int wm    = wave >> 1;                 // 4 waves along m
  const int nWave = blockIdx.y * 128 + wn * 64;
  const int mWave = blockIdx.x * 128 + wm * 32;

  v8f acc[4][2] = {};                          // [n-subtile][m-subtile]

  const int rlane = lane & 15;
  const int khalf = (lane >> 4) << 3;          // ISA A layout: K half per lane group
  size_t rowOff[4];
#pragma unroll
  for (int ns = 0; ns < 4; ++ns)
    rowOff[ns] = (size_t)(nWave + ns * 16 + rlane) * KDIM + khalf;

  for (int k0 = 0; k0 < KDIM; k0 += 32) {
    // A fragments: elems 0..7 <- K khalf+0..7 ; elems 8..15 <- K 16+khalf+0..7
    Vb ahi[4], alo[4];
#pragma unroll
    for (int ns = 0; ns < 4; ++ns) {
      const unsigned short* ph = Xhi + rowOff[ns] + k0;
      const unsigned short* pl = Xlo + rowOff[ns] + k0;
      ahi[ns].h[0] = *(const v8bf*)(ph);
      ahi[ns].h[1] = *(const v8bf*)(ph + 16);
      alo[ns].h[0] = *(const v8bf*)(pl);
      alo[ns].h[1] = *(const v8bf*)(pl + 16);
      __builtin_prefetch(ph + 256, 0, 1);      // x streams from HBM
      __builtin_prefetch(pl + 256, 0, 1);
    }

    // B fragments: lane%16 = M column, K in same order as A (W is m-major)
    const size_t brow = (size_t)(mWave + rlane) * KDIM + k0 + khalf;
    v16bf bhi[2], blo[2];
#pragma unroll
    for (int ms = 0; ms < 2; ++ms) {
      bhi[ms] = __builtin_shufflevector(*(const v8bf*)(WhiU + brow + (size_t)ms * 16 * KDIM), *(const v8bf*)(WhiU + brow + (size_t)ms * 16 * KDIM + 16), 0, 1, 2, 3, 4, 5, 6, 7, 8, 9, 10, 11, 12, 13, 14, 15);
      blo[ms] = __builtin_shufflevector(*(const v8bf*)(WloU + brow + (size_t)ms * 16 * KDIM), *(const v8bf*)(WloU + brow + (size_t)ms * 16 * KDIM + 16), 0, 1, 2, 3, 4, 5, 6, 7, 8, 9, 10, 11, 12, 13, 14, 15);
    }

    // term-major: 8 independent WMMAs between reuses of each accumulator
#pragma unroll
    for (int ns = 0; ns < 4; ++ns)
#pragma unroll
      for (int ms = 0; ms < 2; ++ms)
        acc[ns][ms] = __builtin_amdgcn_wmma_f32_16x16x32_bf16(
            false, ahi[ns].v, false, bhi[ms], (short)0, acc[ns][ms], false, false);
#pragma unroll
    for (int ns = 0; ns < 4; ++ns)
#pragma unroll
      for (int ms = 0; ms < 2; ++ms)
        acc[ns][ms] = __builtin_amdgcn_wmma_f32_16x16x32_bf16(
            false, alo[ns].v, false, bhi[ms], (short)0, acc[ns][ms], false, false);
#pragma unroll
    for (int ns = 0; ns < 4; ++ns)
#pragma unroll
      for (int ms = 0; ms < 2; ++ms)
        acc[ns][ms] = __builtin_amdgcn_wmma_f32_16x16x32_bf16(
            false, ahi[ns].v, false, blo[ms], (short)0, acc[ns][ms], false, false);
  }

  // store D: VGPR g -> row g (lanes 0-15) / g+8 (lanes 16-31); col = lane%16
#pragma unroll
  for (int ns = 0; ns < 4; ++ns) {
    const int nb = nWave + ns * 16 + ((lane >> 4) << 3);
#pragma unroll
    for (int ms = 0; ms < 2; ++ms) {
      const int m = mWave + ms * 16 + rlane;
#pragma unroll
      for (int g = 0; g < 8; ++g)
        y[(size_t)(nb + g) * MDIM + m] = acc[ns][ms][g];
    }
  }
}

// ---------------------------------------------------------------------------
// Fallback GEMM (small ws): in-register f32->bf16 hi/lo conversion of x.
// ---------------------------------------------------------------------------
__global__ __launch_bounds__(256) void qtip_gemm_conv(const float* __restrict__ x,
                                                      const unsigned short* __restrict__ WhiU,
                                                      const unsigned short* __restrict__ WloU,
                                                      float* __restrict__ y)
{
  const int lane  = threadIdx.x & 31;
  const int wave  = threadIdx.x >> 5;
  const int wn    = wave & 3;
  const int wm    = wave >> 2;
  const int nWave = blockIdx.y * 128 + wn * 32;
  const int mWave = blockIdx.x * 64  + wm * 32;

  v8f acc[2][2] = {};
  const int rlane = lane & 15;
  const int khalf = (lane >> 4) << 3;

  for (int k0 = 0; k0 < KDIM; k0 += 32) {
    Vb ahi[2], alo[2];
#pragma unroll
    for (int ns = 0; ns < 2; ++ns) {
      const float* xp = x + (size_t)(nWave + ns * 16 + rlane) * KDIM + k0 + khalf;
#pragma unroll
      for (int j = 0; j < 8; ++j) {
        float f0 = xp[j];
        float f1 = xp[16 + j];
        unsigned short h0 = bf16_rne(f0);
        ahi[ns].u[j] = h0;
        alo[ns].u[j] = bf16_rne(f0 - __uint_as_float((unsigned)h0 << 16));
        unsigned short h1 = bf16_rne(f1);
        ahi[ns].u[8 + j] = h1;
        alo[ns].u[8 + j] = bf16_rne(f1 - __uint_as_float((unsigned)h1 << 16));
      }
    }
    const size_t brow = (size_t)(mWave + rlane) * KDIM + k0 + khalf;
    v16bf bhi[2], blo[2];
#pragma unroll
    for (int ms = 0; ms < 2; ++ms) {
      bhi[ms] = __builtin_shufflevector(*(const v8bf*)(WhiU + brow + (size_t)ms * 16 * KDIM), *(const v8bf*)(WhiU + brow + (size_t)ms * 16 * KDIM + 16), 0, 1, 2, 3, 4, 5, 6, 7, 8, 9, 10, 11, 12, 13, 14, 15);
      blo[ms] = __builtin_shufflevector(*(const v8bf*)(WloU + brow + (size_t)ms * 16 * KDIM), *(const v8bf*)(WloU + brow + (size_t)ms * 16 * KDIM + 16), 0, 1, 2, 3, 4, 5, 6, 7, 8, 9, 10, 11, 12, 13, 14, 15);
    }
#pragma unroll
    for (int ns = 0; ns < 2; ++ns)
#pragma unroll
      for (int ms = 0; ms < 2; ++ms)
        acc[ns][ms] = __builtin_amdgcn_wmma_f32_16x16x32_bf16(
            false, ahi[ns].v, false, bhi[ms], (short)0, acc[ns][ms], false, false);
#pragma unroll
    for (int ns = 0; ns < 2; ++ns)
#pragma unroll
      for (int ms = 0; ms < 2; ++ms)
        acc[ns][ms] = __builtin_amdgcn_wmma_f32_16x16x32_bf16(
            false, alo[ns].v, false, bhi[ms], (short)0, acc[ns][ms], false, false);
#pragma unroll
    for (int ns = 0; ns < 2; ++ns)
#pragma unroll
      for (int ms = 0; ms < 2; ++ms)
        acc[ns][ms] = __builtin_amdgcn_wmma_f32_16x16x32_bf16(
            false, ahi[ns].v, false, blo[ms], (short)0, acc[ns][ms], false, false);
  }

#pragma unroll
  for (int ns = 0; ns < 2; ++ns) {
    const int nb = nWave + ns * 16 + ((lane >> 4) << 3);
#pragma unroll
    for (int ms = 0; ms < 2; ++ms) {
      const int m = mWave + ms * 16 + rlane;
#pragma unroll
      for (int g = 0; g < 8; ++g)
        y[(size_t)(nb + g) * MDIM + m] = acc[ns][ms][g];
    }
  }
}

extern "C" void kernel_launch(void* const* d_in, const int* in_sizes, int n_in,
                              void* d_out, int out_size, void* d_ws, size_t ws_size,
                              hipStream_t stream) {
  const float* inp     = (const float*)d_in[0];   // [4,2048,4096] f32
  const int*   trellis = (const int*)d_in[1];     // [65536,32] i32
  const float* tlut    = (const float*)d_in[2];   // [512,2] f32
  float*       out     = (float*)d_out;           // [8192,4096] f32

  const size_t wElems = (size_t)KDIM * MDIM;      // per W plane
  const size_t xElems = (size_t)NROWS * KDIM;     // per x plane
  unsigned short* Whi = (unsigned short*)d_ws;
  unsigned short* Wlo = Whi + wElems;

  qtip_decode<<<(NT_TILES * 128) / 256, 256, 0, stream>>>(trellis, tlut, Whi, Wlo);

  const size_t need = (2 * wElems + 2 * xElems) * sizeof(unsigned short); // 198MB
  if (ws_size >= need) {
    unsigned short* Xhi = Wlo + wElems;
    unsigned short* Xlo = Xhi + xElems;
    x_split<<<(int)(xElems / 8 / 256), 256, 0, stream>>>(inp, Xhi, Xlo);
    dim3 grid(MDIM / 128, NROWS / 128);           // blockIdx.x = m (L2-friendly order)
    qtip_gemm_split<<<grid, 256, 0, stream>>>(Xhi, Xlo, Whi, Wlo, out);
  } else {
    dim3 grid(MDIM / 64, NROWS / 128);
    qtip_gemm_conv<<<grid, 256, 0, stream>>>(inp, Whi, Wlo, out);
  }
}
